// CausalSelfAttention_41051297415689
// MI455X (gfx1250) — matrix operand order repaired and checked by reading, not compiled, not run on hardware
//
#include <hip/hip_runtime.h>

// ---------------------------------------------------------------------------
// CDNA5 (gfx1250) causal self-attention:
//   - all matmuls on v_wmma_f32_16x16x32_f16 (f32 accumulate)
//   - LDS staging via GLOBAL_LOAD_ASYNC_TO_LDS_B128 (ASYNCcnt), double-buffered
//   - GEMM: 128x256 block tile, 8 waves, 64x64 wave tile (16 WMMA / iter / wave)
// ---------------------------------------------------------------------------

typedef __attribute__((ext_vector_type(16))) _Float16 v16h;
typedef __attribute__((ext_vector_type(8)))  float    v8f;
typedef __attribute__((ext_vector_type(4)))  unsigned int v4u;
typedef __attribute__((ext_vector_type(4)))  _Float16 v4h;

union Frag {            // 16 halfs = one WMMA A/B fragment (8 VGPRs)
  v16h h;
  v4u  u[2];
};

#define HID   2048
#define NH    16
#define HD    128
#define SEQ   2048
#define BATCH 2
#define MTOT  (BATCH * SEQ)   // 4096
#define NQKV  (3 * HID)       // 6144

static __device__ __forceinline__ v8f wmma_f16(v16h a, v16h b, v8f c) {
  return __builtin_amdgcn_wmma_f32_16x16x32_f16(false, a, false, b, (short)0, c,
                                                false, false);
}

// Async global->LDS copy, 16B per lane (tracked by ASYNCcnt, no VGPR bounce).
// lds_off = LDS byte offset (low 32 bits of the generic pointer: the LDS
// aperture truncates flat addresses to addr[31:0], ISA 10.2).
static __device__ __forceinline__ void async_b128(unsigned lds_off,
                                                  const _Float16* g) {
  asm volatile("global_load_async_to_lds_b128 %0, %1, off"
               :
               : "v"(lds_off), "v"(g)
               : "memory");
}
static __device__ __forceinline__ void wait_async0() {
  asm volatile("s_wait_asynccnt 0x0" ::: "memory");
}
static __device__ __forceinline__ unsigned lds_off_of(const void* p) {
  return (unsigned)(unsigned long long)p;
}

// ---------------------------------------------------------------------------
// f32 -> f16 conversion
// ---------------------------------------------------------------------------
__global__ void cvt_f32_to_f16_kernel(const float* __restrict__ src,
                                      _Float16* __restrict__ dst, int n) {
  int i = blockIdx.x * blockDim.x + threadIdx.x;
  if (i < n) dst[i] = (_Float16)src[i];
}

// ---------------------------------------------------------------------------
// Shared GEMM mainloop: C(128x256 block) = A(MxK) * B(NxK)^T, f16 in, f32 acc.
// 256 threads = 8 waves in a 2(M) x 4(N) grid; wave tile 64x64 = 4x4 WMMA tiles.
// Double-buffered LDS fed by async-to-LDS copies.
// ---------------------------------------------------------------------------
__device__ __forceinline__ void gemm_mainloop(const _Float16* __restrict__ A,
                                              const _Float16* __restrict__ Bw,
                                              int K, int m0, int n0,
                                              _Float16* ldsA, _Float16* ldsB,
                                              v8f c[4][4]) {
  const int tid  = threadIdx.x;
  const int lane = tid & 31;
  const int wv   = tid >> 5;
  const int wm   = wv >> 2;        // 0..1
  const int wn   = wv & 3;         // 0..3
  const int h8   = lane >> 4;      // lane half
  const int l16  = lane & 15;

  const int arow = tid & 127;      // A staging row 0..127
  const int aseg = tid >> 7;       // A staging k-segment 0..1 (16 halfs each)

  const unsigned aBase = lds_off_of(ldsA);
  const unsigned bBase = lds_off_of(ldsB);
  const unsigned aOff = (unsigned)(arow * 64 + aseg * 32);  // bytes
  const unsigned bOff = (unsigned)(tid * 64);               // bytes (row = tid)

  auto stage = [&](int k0, int buf) {
    // A tile: 128 rows x 32 halfs; each thread copies 32B of one row
    const _Float16* ga = A + (size_t)(m0 + arow) * K + k0 + aseg * 16;
    unsigned ao = aBase + (unsigned)buf * 8192u + aOff;
    async_b128(ao, ga);
    async_b128(ao + 16, ga + 8);
    // B tile: 256 rows x 32 halfs; each thread copies one full row (64B)
    const _Float16* gb = Bw + (size_t)(n0 + tid) * K + k0;
    unsigned bo = bBase + (unsigned)buf * 16384u + bOff;
    async_b128(bo, gb);
    async_b128(bo + 16, gb + 8);
    async_b128(bo + 32, gb + 16);
    async_b128(bo + 48, gb + 24);
  };

  stage(0, 0);
  for (int k0 = 0; k0 < K; k0 += 32) {
    const int buf = (k0 >> 5) & 1;
    wait_async0();        // async copies for current buffer done (this wave)
    __syncthreads();      // all waves' copies visible; prev buffer fully read
    if (k0 + 32 < K) stage(k0 + 32, buf ^ 1);

    const _Float16* la = ldsA + buf * 4096;
    const _Float16* lb = ldsB + buf * 8192;
    // ---- A fragments: 4 M-subtiles of 16 rows ----
    Frag af[4];
#pragma unroll
    for (int mi = 0; mi < 4; ++mi) {
      int m = wm * 64 + mi * 16 + l16;              // A: row = lane%16
      af[mi].u[0] = *(const v4u*)&la[m * 32 + 8 * h8];        // K 0..7 / 8..15
      af[mi].u[1] = *(const v4u*)&la[m * 32 + 16 + 8 * h8];   // K 16..23 / 24..31
    }
    // ---- B fragments: 4 N-subtiles of 16 cols ----
    Frag bf[4];
#pragma unroll
    for (int ni = 0; ni < 4; ++ni) {
      int n = wn * 64 + ni * 16 + l16;              // B: col = lane%16
      bf[ni].u[0] = *(const v4u*)&lb[n * 32 + 8 * h8];        // K 0..7 / 8..15
      bf[ni].u[1] = *(const v4u*)&lb[n * 32 + 16 + 8 * h8];   // K 16..23 / 24..31
    }
#pragma unroll
    for (int mi = 0; mi < 4; ++mi)
#pragma unroll
      for (int ni = 0; ni < 4; ++ni)
        c[mi][ni] = wmma_f16(af[mi].h, bf[ni].h, c[mi][ni]);
  }
}

// ---------------------------------------------------------------------------
// GEMM 1: qkv = x @ w_qkv^T + b_qkv, scattered into per-head Q/K/V f16 buffers
// ---------------------------------------------------------------------------
__global__ __launch_bounds__(256) void gemm_qkv_kernel(
    const _Float16* __restrict__ Xh, const _Float16* __restrict__ Wh,
    const float* __restrict__ bias, _Float16* __restrict__ Q,
    _Float16* __restrict__ Kc, _Float16* __restrict__ V) {
  __shared__ _Float16 ldsA[2 * 128 * 32];
  __shared__ _Float16 ldsB[2 * 256 * 32];
  const int m0 = blockIdx.y * 128;
  const int n0 = blockIdx.x * 256;
  v8f c[4][4] = {};
  gemm_mainloop(Xh, Wh, HID, m0, n0, ldsA, ldsB, c);

  const int lane = threadIdx.x & 31, wv = threadIdx.x >> 5;
  const int wm = wv >> 2, wn = wv & 3, h8 = lane >> 4, l16 = lane & 15;
#pragma unroll
  for (int mi = 0; mi < 4; ++mi) {
#pragma unroll
    for (int ni = 0; ni < 4; ++ni) {
      int gn = n0 + wn * 64 + ni * 16 + l16;
      float bv = bias[gn];
      int comp = gn >> 11;          // 0=q 1=k 2=v
      int head = (gn >> 7) & 15;
      int d    = gn & 127;
      _Float16* dst = (comp == 0) ? Q : (comp == 1 ? Kc : V);
#pragma unroll
      for (int r = 0; r < 8; ++r) {
        int m = m0 + wm * 64 + mi * 16 + r + 8 * h8;   // C: row = vgpr + 8*half
        int b = m >> 11, s = m & 2047;
        size_t idx = ((size_t)((b * NH + head) * SEQ + s)) * HD + d;
        dst[idx] = (_Float16)(c[mi][ni][r] + bv);
      }
    }
  }
}

// ---------------------------------------------------------------------------
// RoPE applied in place to Q and K ([b,h,s,d] f16)
// ---------------------------------------------------------------------------
__global__ void rope_kernel(_Float16* Q, _Float16* Kc) {
  int idx = blockIdx.x * blockDim.x + threadIdx.x;   // BATCH*NH*SEQ*(HD/2)
  int i  = idx & 63;
  int s  = (idx >> 6) & (SEQ - 1);
  int bh = idx >> 17;
  float inv = __powf(10000.f, -((float)(2 * i) / (float)HD));
  float ang = (float)s * inv;
  float sn, cs;
  __sincosf(ang, &sn, &cs);
  size_t base = ((size_t)bh * SEQ + s) * HD + 2 * i;
  float qe = (float)Q[base], qo = (float)Q[base + 1];
  Q[base]     = (_Float16)(qe * cs - qo * sn);
  Q[base + 1] = (_Float16)(qe * sn + qo * cs);
  float ke = (float)Kc[base], ko = (float)Kc[base + 1];
  Kc[base]     = (_Float16)(ke * cs - ko * sn);
  Kc[base + 1] = (_Float16)(ke * sn + ko * cs);
}

// ---------------------------------------------------------------------------
// Flash attention: block = 4 waves x 16 q-rows, streams 32-key K/V chunks.
// Double-buffered staging: K via async-to-LDS, V transposed through registers
// with packed b64 LDS stores (4 k-rows x 8 d per thread).
// ---------------------------------------------------------------------------
__global__ __launch_bounds__(128) void attn_kernel(
    const _Float16* __restrict__ Q, const _Float16* __restrict__ Kc,
    const _Float16* __restrict__ V, _Float16* __restrict__ Y) {
  __shared__ _Float16 ldsK[2 * 32 * 128];   // [buf][k][d]
  __shared__ _Float16 ldsV[2 * 128 * 32];   // [buf][d][k]  (transposed)
  __shared__ _Float16 ldsP[4][16 * 32];     // per-wave P restage

  const int tid  = threadIdx.x;
  const int lane = tid & 31, wv = tid >> 5;
  const int h8 = lane >> 4, l16 = lane & 15;
  const int qblk = blockIdx.x, head = blockIdx.y, b = blockIdx.z;
  const int bh = b * NH + head;
  const _Float16* Qb = Q + (size_t)bh * SEQ * HD;
  const _Float16* Kb = Kc + (size_t)bh * SEQ * HD;
  const _Float16* Vb = V + (size_t)bh * SEQ * HD;
  const int q0 = qblk * 64 + wv * 16;

  // Q fragments (held in registers for the whole sweep)
  Frag qf[4];
  {
    const _Float16* qrow = Qb + (size_t)(q0 + l16) * HD;
#pragma unroll
    for (int kk = 0; kk < 4; ++kk) {
      qf[kk].u[0] = *(const v4u*)&qrow[kk * 32 + 8 * h8];
      qf[kk].u[1] = *(const v4u*)&qrow[kk * 32 + 16 + 8 * h8];
    }
  }

  v8f o[8] = {};
  float mrow[8], lrow[8];
#pragma unroll
  for (int r = 0; r < 8; ++r) { mrow[r] = -1e30f; lrow[r] = 0.f; }
  const float scale = 0.08838834764831845f;   // 1/sqrt(128)

  const unsigned kBase = lds_off_of(ldsK);
  const int sk = tid & 31, sseg = tid >> 5;   // K stage: 32 rows x 4 segs
  const int vk4 = (tid & 7) * 4;              // V stage: 4 k-rows ...
  const int vdseg = tid >> 3;                 // ... x 8 d per thread

  auto stageKV = [&](int j0s, int buf) {
    // K chunk [32][128] via async-to-LDS (4 x 16B per thread)
    const _Float16* ks = &Kb[(size_t)(j0s + sk) * HD + sseg * 32];
    unsigned ko = kBase + (unsigned)buf * 8192u +
                  (unsigned)(sk * 128 + sseg * 32) * 2u;
    async_b128(ko, ks);
    async_b128(ko + 16, ks + 8);
    async_b128(ko + 32, ks + 16);
    async_b128(ko + 48, ks + 24);
    // V chunk transposed [128][32]: 4 rows x 8 d, packed b64 stores
    union { v4u u; _Float16 h[8]; } r[4];
#pragma unroll
    for (int kk = 0; kk < 4; ++kk)
      r[kk].u = *(const v4u*)&Vb[(size_t)(j0s + vk4 + kk) * HD + vdseg * 8];
    _Float16* vb = ldsV + buf * 4096;
#pragma unroll
    for (int i = 0; i < 8; ++i) {
      v4h pack = {r[0].h[i], r[1].h[i], r[2].h[i], r[3].h[i]};
      *(v4h*)&vb[(vdseg * 8 + i) * 32 + vk4] = pack;
    }
  };

  const int jend = qblk * 64 + 64;
  stageKV(0, 0);
  for (int j0 = 0; j0 < jend; j0 += 32) {
    const int buf = (j0 >> 5) & 1;
    wait_async0();
    __syncthreads();     // staging visible; prev buffer fully consumed
    if (j0 + 32 < jend) stageKV(j0 + 32, buf ^ 1);
    const _Float16* lK = ldsK + buf * 4096;
    const _Float16* lV = ldsV + buf * 4096;

    if (j0 <= q0 + 15) {                     // wave-uniform: EXEC stays full
      // ---- scores: two 16x16 tiles over 32 keys, K-dim = 128 ----
      v8f sc[2];
#pragma unroll
      for (int kt = 0; kt < 2; ++kt) {
        v8f acc = {};
        int krow = kt * 16 + l16;            // B: col = key (lane%16)
#pragma unroll
        for (int kk = 0; kk < 4; ++kk) {
          Frag bfr;
          bfr.u[0] = *(const v4u*)&lK[krow * 128 + kk * 32 + 8 * h8];
          bfr.u[1] = *(const v4u*)&lK[krow * 128 + kk * 32 + 16 + 8 * h8];
          acc = wmma_f16(qf[kk].h, bfr.h, acc);
        }
        sc[kt] = acc;
      }
      // ---- online softmax (row = q0 + r + 8*h8, one row per VGPR per half) ----
      float corr[8];
#pragma unroll
      for (int r = 0; r < 8; ++r) {
        int row = q0 + r + 8 * h8;
        float v0 = (j0 + l16 <= row)      ? sc[0][r] * scale : -1e30f;
        float v1 = (j0 + 16 + l16 <= row) ? sc[1][r] * scale : -1e30f;
        float t = fmaxf(v0, v1);
        t = fmaxf(t, __shfl_xor(t, 8, 16));
        t = fmaxf(t, __shfl_xor(t, 4, 16));
        t = fmaxf(t, __shfl_xor(t, 2, 16));
        t = fmaxf(t, __shfl_xor(t, 1, 16));
        float nm = fmaxf(mrow[r], t);
        corr[r] = __expf(mrow[r] - nm);
        float p0 = __expf(v0 - nm), p1 = __expf(v1 - nm);
        float rs = p0 + p1;
        rs += __shfl_xor(rs, 8, 16);
        rs += __shfl_xor(rs, 4, 16);
        rs += __shfl_xor(rs, 2, 16);
        rs += __shfl_xor(rs, 1, 16);
        lrow[r] = lrow[r] * corr[r] + rs;
        mrow[r] = nm;
        // restage P (C-layout -> LDS row-major) for A-fragment reload
        ldsP[wv][(r + 8 * h8) * 32 + l16]      = (_Float16)p0;
        ldsP[wv][(r + 8 * h8) * 32 + 16 + l16] = (_Float16)p1;
      }
#pragma unroll
      for (int nt = 0; nt < 8; ++nt)
#pragma unroll
        for (int r = 0; r < 8; ++r) o[nt][r] *= corr[r];

      asm volatile("" ::: "memory");   // keep DS stores before DS loads (in-order HW)
      Frag pf;
      pf.u[0] = *(const v4u*)&ldsP[wv][l16 * 32 + 8 * h8];
      pf.u[1] = *(const v4u*)&ldsP[wv][l16 * 32 + 16 + 8 * h8];
      // ---- O += P(16x32) @ V(32x128) ----
#pragma unroll
      for (int nt = 0; nt < 8; ++nt) {
        int d = nt * 16 + l16;
        Frag vf;
        vf.u[0] = *(const v4u*)&lV[d * 32 + 8 * h8];
        vf.u[1] = *(const v4u*)&lV[d * 32 + 16 + 8 * h8];
        o[nt] = wmma_f16(pf.h, vf.h, o[nt]);
      }
    }
  }

  // epilogue: normalize, write Y[b][s][head*128+d] as f16
#pragma unroll
  for (int nt = 0; nt < 8; ++nt) {
#pragma unroll
    for (int r = 0; r < 8; ++r) {
      int row = q0 + r + 8 * h8;
      float val = o[nt][r] / lrow[r];
      size_t idx = ((size_t)(b * SEQ + row)) * HID + head * HD + nt * 16 + l16;
      Y[idx] = (_Float16)val;
    }
  }
}

// ---------------------------------------------------------------------------
// GEMM 3: out = Y @ w_out^T + b_out  (f32 output)
// ---------------------------------------------------------------------------
__global__ __launch_bounds__(256) void gemm_out_kernel(
    const _Float16* __restrict__ Yh, const _Float16* __restrict__ Wh,
    const float* __restrict__ bias, float* __restrict__ out) {
  __shared__ _Float16 ldsA[2 * 128 * 32];
  __shared__ _Float16 ldsB[2 * 256 * 32];
  const int m0 = blockIdx.y * 128;
  const int n0 = blockIdx.x * 256;
  v8f c[4][4] = {};
  gemm_mainloop(Yh, Wh, HID, m0, n0, ldsA, ldsB, c);

  const int lane = threadIdx.x & 31, wv = threadIdx.x >> 5;
  const int wm = wv >> 2, wn = wv & 3, h8 = lane >> 4, l16 = lane & 15;
#pragma unroll
  for (int mi = 0; mi < 4; ++mi) {
#pragma unroll
    for (int ni = 0; ni < 4; ++ni) {
      int gn = n0 + wn * 64 + ni * 16 + l16;
      float bv = bias[gn];
#pragma unroll
      for (int r = 0; r < 8; ++r) {
        int m = m0 + wm * 64 + mi * 16 + r + 8 * h8;
        out[(size_t)m * HID + gn] = c[mi][ni][r] + bv;
      }
    }
  }
}

// ---------------------------------------------------------------------------
extern "C" void kernel_launch(void* const* d_in, const int* in_sizes, int n_in,
                              void* d_out, int out_size, void* d_ws,
                              size_t ws_size, hipStream_t stream) {
  const float* x     = (const float*)d_in[0];
  const float* w_qkv = (const float*)d_in[1];
  const float* b_qkv = (const float*)d_in[2];
  const float* w_out = (const float*)d_in[3];
  const float* b_out = (const float*)d_in[4];
  float* out = (float*)d_out;

  char* p = (char*)d_ws;
  _Float16* Xh    = (_Float16*)p; p += (size_t)MTOT * HID * 2;
  _Float16* Wqkvh = (_Float16*)p; p += (size_t)NQKV * HID * 2;
  _Float16* Wouth = (_Float16*)p; p += (size_t)HID * HID * 2;
  _Float16* Qh    = (_Float16*)p; p += (size_t)BATCH * NH * SEQ * HD * 2;
  _Float16* Kh    = (_Float16*)p; p += (size_t)BATCH * NH * SEQ * HD * 2;
  _Float16* Vh    = (_Float16*)p; p += (size_t)BATCH * NH * SEQ * HD * 2;
  _Float16* Yh    = (_Float16*)p; p += (size_t)MTOT * HID * 2;

  int nx = MTOT * HID, nw = NQKV * HID, no = HID * HID;
  cvt_f32_to_f16_kernel<<<nx / 256, 256, 0, stream>>>(x, Xh, nx);
  cvt_f32_to_f16_kernel<<<nw / 256, 256, 0, stream>>>(w_qkv, Wqkvh, nw);
  cvt_f32_to_f16_kernel<<<no / 256, 256, 0, stream>>>(w_out, Wouth, no);

  gemm_qkv_kernel<<<dim3(NQKV / 256, MTOT / 128), 256, 0, stream>>>(
      Xh, Wqkvh, b_qkv, Qh, Kh, Vh);

  int nrope = BATCH * NH * SEQ * (HD / 2);
  rope_kernel<<<nrope / 256, 256, 0, stream>>>(Qh, Kh);

  attn_kernel<<<dim3(SEQ / 64, NH, BATCH), 128, 0, stream>>>(Qh, Kh, Vh, Yh);

  gemm_out_kernel<<<dim3(HID / 256, MTOT / 128), 256, 0, stream>>>(
      Yh, Wouth, b_out, out);
}
